// Attention_10479720202400
// MI455X (gfx1250) — compile-verified, matrix operand order checked by reading, not run on hardware
//
#include <hip/hip_runtime.h>
#include <hip/hip_bf16.h>
#include <stdint.h>

typedef __attribute__((ext_vector_type(16))) __bf16 bf16x16;
typedef __attribute__((ext_vector_type(2)))  __bf16 bf16x2;
typedef __attribute__((ext_vector_type(8)))  float  f32x8;
typedef __attribute__((ext_vector_type(4)))  int    v4i;

union FragB { bf16x16 v; unsigned u[8]; };

// ---- f32 -> bf16 helpers (prefer v_cvt_pk_bf16_f32) ------------------------
__device__ __forceinline__ unsigned bfbits(float f) {
  unsigned a = __float_as_uint(f);
  return (a + 0x7FFFu + ((a >> 16) & 1u)) >> 16;   // RNE f32 -> bf16
}
__device__ __forceinline__ unsigned pk2(float lo, float hi) {
#if defined(__has_builtin) && __has_builtin(__builtin_amdgcn_cvt_pk_bf16_f32)
  union { bf16x2 v; unsigned u; } c;
  c.v = __builtin_amdgcn_cvt_pk_bf16_f32(lo, hi);
  return c.u;
#else
  return (bfbits(lo) & 0xFFFFu) | (bfbits(hi) << 16);
#endif
}

// ---- async global->LDS (CDNA5), guarded with plain-load fallback -----------
#if defined(__has_builtin) && __has_builtin(__builtin_amdgcn_global_load_async_to_lds_b128)
#define HAVE_ASYNC_LDS 1
#define AS1CAST(p) ((__attribute__((address_space(1))) v4i*)(p))
#define AS3CAST(p) ((__attribute__((address_space(3))) v4i*)(p))
#else
#define HAVE_ASYNC_LDS 0
#endif

__device__ __forceinline__ void wait_async0() {
#if defined(__has_builtin) && __has_builtin(__builtin_amdgcn_s_wait_asynccnt)
  __builtin_amdgcn_s_wait_asynccnt(0);
#else
  asm volatile("s_wait_asynccnt 0x0" ::: "memory");
#endif
}

__device__ __forceinline__ void copy16_g2l(const void* g, unsigned* l) {
#if HAVE_ASYNC_LDS
  __builtin_amdgcn_global_load_async_to_lds_b128(AS1CAST(g), AS3CAST(l), 0, 0);
#else
  uint4 q = *(const uint4*)g;
  l[0] = q.x; l[1] = q.y; l[2] = q.z; l[3] = q.w;
#endif
}

// ---------------------------------------------------------------------------
// Fused bias: biasf[h][nm(pad 2404)] = rpb[rel[nm]*16 + h]; pad keeps rows
// 16B-aligned for async B128 staging.
// ---------------------------------------------------------------------------
__global__ __launch_bounds__(256) void bias_fuse(
    const float* __restrict__ rpb, const int* __restrict__ rel,
    float* __restrict__ biasf) {
  int i = blockIdx.x * 256 + threadIdx.x;           // over 16*2401
  if (i < 16 * 2401) {
    int h = i / 2401, nm = i % 2401;
    biasf[h * 2404 + nm] = rpb[rel[nm] * 16 + h];
  }
}

// ---------------------------------------------------------------------------
// Tiled bf16 WMMA GEMM: Out[M,N] = A[M,512] @ W[N,512]^T + bias[N]
// Block 256 thr (8 waves), tile 128(M) x 64(N), K-tile 64, double-buffered
// LDS (1 barrier/stage). Wave computes 32x32 via 2x2 v_wmma per K-half.
// LDS row stride = 36 dwords: 16B-aligned rows (async B128 ok) AND 36 mod 64
// steps 16 distinct banks across 16 fragment rows (conflict-free reads).
// ---------------------------------------------------------------------------
template <bool A_BF16, bool OUT_BF16, bool QSCALE>
__global__ __launch_bounds__(256) void gemm_wmma(
    const void* __restrict__ Ap, const float* __restrict__ W,
    const float* __restrict__ bias, void* __restrict__ Out, int N) {
  constexpr int K = 512;
  constexpr int RS = 36;                       // dwords per 64-bf16 row
  __shared__ __attribute__((aligned(16))) unsigned As[2][128 * RS];
  __shared__ __attribute__((aligned(16))) unsigned Bs[2][64 * RS];
  const int t = threadIdx.x;
  const int lane = t & 31, lr = lane & 15, half = lane >> 4;
  const int wid = t >> 5, waveM = wid & 3, waveN = wid >> 2;
  const size_t blockM = (size_t)blockIdx.x * 128;
  const int blockN = blockIdx.y * 64;

  f32x8 acc[2][2] = {};

  auto loadA = [&](int k0, int buf) {
    if (A_BF16) {
      const unsigned short* Ab = (const unsigned short*)Ap;
#pragma unroll
      for (int p = 0; p < 4; ++p) {
        int rl = p * 32 + (t >> 3);
        int qc = t & 7;                        // 8 bf16 chunks per 64-wide row
        copy16_g2l(Ab + (blockM + rl) * K + k0 + qc * 8,
                   &As[buf][rl * RS + qc * 4]);
      }
    } else {
      const float* Af = (const float*)Ap;
#pragma unroll
      for (int p = 0; p < 8; ++p) {
        int rl = p * 16 + (t >> 4);
        int fc = t & 15;                       // 16 float4 per 64-wide row
        float4 f = *(const float4*)(Af + (blockM + rl) * K + k0 + fc * 4);
        As[buf][rl * RS + fc * 2 + 0] = pk2(f.x, f.y);
        As[buf][rl * RS + fc * 2 + 1] = pk2(f.z, f.w);
      }
    }
  };
  auto loadB = [&](int k0, int buf) {
#pragma unroll
    for (int p = 0; p < 4; ++p) {
      int nl = p * 16 + (t >> 4);
      int fc = t & 15;
      float4 f = *(const float4*)(W + (size_t)(blockN + nl) * K + k0 + fc * 4);
      Bs[buf][nl * RS + fc * 2 + 0] = pk2(f.x, f.y);
      Bs[buf][nl * RS + fc * 2 + 1] = pk2(f.z, f.w);
    }
  };
  auto compute = [&](int buf) {
#pragma unroll
    for (int kh = 0; kh < 2; ++kh) {
      FragB fa[2], fb[2];
#pragma unroll
      for (int mt = 0; mt < 2; ++mt) {
        int row = waveM * 32 + mt * 16 + lr;
        int base = row * RS + kh * 16 + half * 4;
#pragma unroll
        for (int v = 0; v < 4; ++v) {
          fa[mt].u[v] = As[buf][base + v];
          fa[mt].u[v + 4] = As[buf][base + 8 + v];
        }
      }
#pragma unroll
      for (int nt = 0; nt < 2; ++nt) {
        int col = waveN * 32 + nt * 16 + lr;
        int base = col * RS + kh * 16 + half * 4;
#pragma unroll
        for (int v = 0; v < 4; ++v) {
          fb[nt].u[v] = Bs[buf][base + v];
          fb[nt].u[v + 4] = Bs[buf][base + 8 + v];
        }
      }
#pragma unroll
      for (int mt = 0; mt < 2; ++mt)
#pragma unroll
        for (int nt = 0; nt < 2; ++nt)
          acc[mt][nt] = __builtin_amdgcn_wmma_f32_16x16x32_bf16(
              false, fa[mt].v, false, fb[nt].v, (short)0, acc[mt][nt], false,
              false);
    }
  };

  loadA(0, 0);
  loadB(0, 0);
  if (A_BF16 && HAVE_ASYNC_LDS) wait_async0();
  __syncthreads();
  for (int s = 0; s < 8; ++s) {                // K = 8 stages of 64
    int buf = s & 1;
    if (s < 7) { loadA((s + 1) * 64, buf ^ 1); loadB((s + 1) * 64, buf ^ 1); }
    compute(buf);
    if (A_BF16 && HAVE_ASYNC_LDS && s < 7) wait_async0();
    __syncthreads();
  }

#pragma unroll
  for (int mt = 0; mt < 2; ++mt)
#pragma unroll
    for (int nt = 0; nt < 2; ++nt) {
      int coln = blockN + waveN * 32 + nt * 16 + lr;
      float bv = bias[coln];
#pragma unroll
      for (int r = 0; r < 8; ++r) {
        size_t row = blockM + waveM * 32 + mt * 16 + r + half * 8;
        float val = acc[mt][nt][r] + bv;
        if (QSCALE) {
          if (coln < 512) val *= 0.17677669529663689f;   // 1/sqrt(32)
        }
        if (OUT_BF16)
          ((unsigned short*)Out)[row * N + coln] = (unsigned short)bfbits(val);
        else
          ((float*)Out)[row * N + coln] = val;
      }
    }
}

// ---------------------------------------------------------------------------
// Window attention: 1 wave per (window b, head h); 4 waves/block SHARE one
// head so the 9.6KB bias slice is staged to LDS once (async) per block.
// qkv: [token][1536] bf16, cols {q|k|v} x h x d, q pre-scaled by 1/sqrt(32).
// ---------------------------------------------------------------------------
__global__ __launch_bounds__(128) void attn_wmma(
    const unsigned short* __restrict__ qkv, const float* __restrict__ biasf,
    unsigned short* __restrict__ outp) {
  __shared__ float S[4][49 * 51];                       // per-wave scores
  __shared__ __attribute__((aligned(16))) float Bsh[2404];  // head bias slice
  const int t = threadIdx.x;
  const int wid = t >> 5;
  const int lane = t & 31, lr = lane & 15, half = lane >> 4;
  const int kb = half * 8;
  const int h = blockIdx.x & 15;
  const int b = ((blockIdx.x >> 4) << 2) + wid;
  float* Sw = S[wid];

  // ---- stage this head's bias slice into LDS (601 x 16B chunks) ----
#if HAVE_ASYNC_LDS
  for (int i = t; i < 601; i += 128)
    copy16_g2l(biasf + (size_t)h * 2404 + i * 4, (unsigned*)&Bsh[i * 4]);
#else
  for (int i = t; i < 2404; i += 128) Bsh[i] = biasf[(size_t)h * 2404 + i];
#endif

  // ---- Q fragments (A-layout: lane->row, VGPR->packed K pairs) ----
  FragB aq[4];
#pragma unroll
  for (int mt = 0; mt < 4; ++mt) {
    int tok = mt * 16 + lr;
    if (tok < 49) {
      const unsigned* p =
          (const unsigned*)(qkv + (size_t)(b * 49 + tok) * 1536 + h * 32);
      int bq = half * 4;
#pragma unroll
      for (int v = 0; v < 4; ++v) {
        aq[mt].u[v] = p[bq + v];
        aq[mt].u[v + 4] = p[8 + bq + v];
      }
    } else {
#pragma unroll
      for (int v = 0; v < 8; ++v) aq[mt].u[v] = 0u;
    }
  }
#if HAVE_ASYNC_LDS
  wait_async0();
#endif
  __syncthreads();                                      // bias slice visible

  // ---- S = Q K^T + bias, streamed over key tiles ----
  for (int jt = 0; jt < 4; ++jt) {
    FragB bk;
    {
      int tok = jt * 16 + lr;
      if (tok < 49) {
        const unsigned* p = (const unsigned*)(qkv +
            (size_t)(b * 49 + tok) * 1536 + 512 + h * 32);
        int bq = half * 4;
#pragma unroll
        for (int v = 0; v < 4; ++v) {
          bk.u[v] = p[bq + v];
          bk.u[v + 4] = p[8 + bq + v];
        }
      } else {
#pragma unroll
        for (int v = 0; v < 8; ++v) bk.u[v] = 0u;
      }
    }
#pragma unroll
    for (int mt = 0; mt < 4; ++mt) {
      f32x8 sc = {};
      sc = __builtin_amdgcn_wmma_f32_16x16x32_bf16(
          false, aq[mt].v, false, bk.v, (short)0, sc, false, false);
      int j = jt * 16 + lr;
#pragma unroll
      for (int r = 0; r < 8; ++r) {
        int n = mt * 16 + r + half * 8;
        if (n < 49 && j < 49)
          Sw[n * 51 + j] = sc[r] + Bsh[n * 49 + j];
      }
    }
  }
  __syncthreads();

  // ---- f32 row softmax (lane = row; stride 51 is bank-conflict-free) ----
#pragma unroll
  for (int rb = 0; rb < 64; rb += 32) {
    int r = rb + lane;
    if (r < 49) {
      float* row = Sw + r * 51;
      float m = -1e30f;
      for (int c = 0; c < 49; ++c) m = fmaxf(m, row[c]);
      float s = 0.f;
      for (int c = 0; c < 49; ++c) {
        float e = __expf(row[c] - m);
        row[c] = e;
        s += e;
      }
      float inv = 1.0f / s;
      for (int c = 0; c < 49; ++c) row[c] *= inv;
    }
  }
  __syncthreads();

  // ---- O = P V ----
  FragB ap[4][2];
#pragma unroll
  for (int mt = 0; mt < 4; ++mt) {
    int row = mt * 16 + lr;
#pragma unroll
    for (int kt = 0; kt < 2; ++kt) {
#pragma unroll
      for (int v = 0; v < 8; ++v) {
        int kk = kt * 32 + ((v < 4) ? (kb + 2 * v) : (16 + kb + 2 * (v - 4)));
        float lo = (row < 49 && kk < 49) ? Sw[row * 51 + kk] : 0.f;
        float hi = (row < 49 && kk + 1 < 49) ? Sw[row * 51 + kk + 1] : 0.f;
        ap[mt][kt].u[v] = pk2(lo, hi);
      }
    }
  }
#pragma unroll
  for (int nt = 0; nt < 2; ++nt) {
    FragB bv[2];
#pragma unroll
    for (int kt = 0; kt < 2; ++kt) {
      int d = nt * 16 + lr;
#pragma unroll
      for (int v = 0; v < 8; ++v) {
        int j = kt * 32 + ((v < 4) ? (kb + 2 * v) : (16 + kb + 2 * (v - 4)));
        unsigned lo = (j < 49)
            ? (unsigned)qkv[(size_t)(b * 49 + j) * 1536 + 1024 + h * 32 + d]
            : 0u;
        unsigned hi = (j + 1 < 49)
            ? (unsigned)qkv[(size_t)(b * 49 + j + 1) * 1536 + 1024 + h * 32 + d]
            : 0u;
        bv[kt].u[v] = lo | (hi << 16);
      }
    }
#pragma unroll
    for (int mt = 0; mt < 4; ++mt) {
      f32x8 o = {};
      o = __builtin_amdgcn_wmma_f32_16x16x32_bf16(
          false, ap[mt][0].v, false, bv[0].v, (short)0, o, false, false);
      o = __builtin_amdgcn_wmma_f32_16x16x32_bf16(
          false, ap[mt][1].v, false, bv[1].v, (short)0, o, false, false);
      int d = nt * 16 + lr;
#pragma unroll
      for (int r = 0; r < 8; ++r) {
        int n = mt * 16 + r + half * 8;
        if (n < 49)
          outp[(size_t)(b * 49 + n) * 512 + h * 32 + d] =
              (unsigned short)bfbits(o[r]);
      }
    }
  }
}

// ---------------------------------------------------------------------------
extern "C" void kernel_launch(void* const* d_in, const int* in_sizes, int n_in,
                              void* d_out, int out_size, void* d_ws,
                              size_t ws_size, hipStream_t stream) {
  (void)in_sizes; (void)n_in; (void)out_size; (void)ws_size;
  const float* x      = (const float*)d_in[0];
  const float* qkv_w  = (const float*)d_in[1];
  const float* qkv_b  = (const float*)d_in[2];
  const float* proj_w = (const float*)d_in[3];
  const float* proj_b = (const float*)d_in[4];
  const float* rpb    = (const float*)d_in[5];
  const int*   rel    = (const int*)d_in[6];

  char* ws = (char*)d_ws;
  unsigned short* qkvbuf  = (unsigned short*)ws;                    // 100352*1536*2 = 308,281,344 B
  unsigned short* attnbuf = (unsigned short*)(ws + 308281344ull);   // 100352*512*2  = 102,760,448 B
  float*          biasf   = (float*)(ws + 308281344ull + 102760448ull);  // 16*2404*4

  bias_fuse<<<dim3((16 * 2401 + 255) / 256), dim3(256), 0, stream>>>(
      rpb, rel, biasf);
  // QKV: M=100352, N=1536 -> grid 784 x 24
  gemm_wmma<false, true, true><<<dim3(784, 24), dim3(256), 0, stream>>>(
      x, qkv_w, qkv_b, qkvbuf, 1536);
  // Attention: 2048 windows * 16 heads; 4 waves/block share one head
  attn_wmma<<<dim3(8192), dim3(128), 0, stream>>>(qkvbuf, biasf, attnbuf);
  // Proj: M=100352, N=512 -> grid 784 x 8, f32 output
  gemm_wmma<true, false, false><<<dim3(784, 8), dim3(256), 0, stream>>>(
      attnbuf, proj_w, proj_b, d_out, 512);
}
